// EFN_4123168604885
// MI455X (gfx1250) — matrix operand order repaired and checked by reading, not compiled, not run on hardware
//
#include <hip/hip_runtime.h>
#include <stdint.h>

typedef _Float16 v16h __attribute__((ext_vector_type(16)));
typedef _Float16 v8h  __attribute__((ext_vector_type(8)));
typedef float    v8f  __attribute__((ext_vector_type(8)));

#define N_NODES 100000
#define N_EDGES 3200000
#define F_X 24
#define D_IN 32
#define H_DIM 64
#define F_OUT 32

// Hardware FP32 atomic add, SADDR (GVS) form:
//   mem = SGPR64 base + zext(VGPR32 offset) + imm
// Non-returning -> STOREcnt tracked; no CAS loop possible.
__device__ __forceinline__ void gfadd(unsigned voff, float v,
                                      unsigned long long sbase, int imm) {
    asm volatile("global_atomic_add_f32 %0, %1, %2 offset:%3"
                 :: "v"(voff), "v"(v), "s"(sbase), "i"(imm)
                 : "memory");
}

// Single-instruction ReLU that the compiler fully understands (so the
// WMMA->VALU data-hazard recognizer still protects the D-register read):
// med3(x, 0, +inf) == clamp(x, 0, inf) == relu(x) -> one v_med3_num_f32.
__device__ __forceinline__ float relu1(float x) {
    return __builtin_amdgcn_fmed3f(x, 0.0f, __builtin_inff());
}

// ---------------------------------------------------------------------------
// 0) zero acc + deg (contiguous in workspace)
__global__ void efn_zero(float* __restrict__ p, long long n) {
    long long i = (long long)blockIdx.x * blockDim.x + threadIdx.x;
    long long stride = (long long)gridDim.x * blockDim.x;
    for (; i < n; i += stride) p[i] = 0.0f;
}

// 1) h16[n][k] = f16( k<24 ? x[n][k] : scalars[k-24] )
__global__ void efn_prep_nodes(const float* __restrict__ x,
                               const float* __restrict__ scalars,
                               _Float16* __restrict__ h16) {
    long long total = (long long)N_NODES * D_IN;
    long long i = (long long)blockIdx.x * blockDim.x + threadIdx.x;
    long long stride = (long long)gridDim.x * blockDim.x;
    for (; i < total; i += stride) {
        int node = (int)(i >> 5);
        int k    = (int)(i & 31);
        float v = (k < F_X) ? x[(size_t)node * F_X + k] : scalars[k - F_X];
        h16[i] = (_Float16)v;
    }
}

// 2) Pre-shuffle W1 (32x64) and W2 (64x32) into WMMA B-operand layout:
//    lane l holds column n=l&15, K rows (l>>4)*8 .. +7, then 16+(l>>4)*8 .. +7.
__global__ void efn_prep_weights(const float* __restrict__ W1,
                                 const float* __restrict__ W2,
                                 _Float16* __restrict__ w1B,
                                 _Float16* __restrict__ w2B) {
    int t = threadIdx.x;          // 128 threads
    int lane = t & 31;
    int g = t >> 5;               // 0..3
    int n = lane & 15, hi = lane >> 4;
    #pragma unroll
    for (int e = 0; e < 16; ++e)
        w1B[(g * 32 + lane) * 16 + e] =
            (_Float16)W1[(hi * 8 + e + (e & 8)) * H_DIM + g * 16 + n];
    int s = g >> 1, j = g & 1;
    #pragma unroll
    for (int e = 0; e < 16; ++e)
        w2B[(g * 32 + lane) * 16 + e] =
            (_Float16)W2[(s * 32 + hi * 8 + e + (e & 8)) * F_OUT + j * 16 + n];
}

// 3a) Degree histogram: deg[dst[e]] += 1, one thread per edge.
//     Load only the low dword of each int64 id (values < 2^31).
__global__ void efn_deg(const long long* __restrict__ dst,
                        float* __restrict__ deg, int n) {
    unsigned long long degB = (unsigned long long)(size_t)deg;
    const char* dstB = (const char*)dst;
    int i = blockIdx.x * blockDim.x + threadIdx.x;
    int stride = gridDim.x * blockDim.x;
    for (; i < n; i += stride) {
        int d = *(const int*)(dstB + (size_t)i * 8u);
        gfadd((unsigned)d << 2, 1.0f, degB, 0);
    }
}

// 3b) Per-edge layer-1: one wave = 16 edges. Branch-free hot loop with
//     index loads software-pipelined one iteration ahead.
//     acc[dst] += relu(h[src]·W1 + b1)
__global__ void efn_edge(const _Float16* __restrict__ h16,
                         const long long* __restrict__ src,
                         const long long* __restrict__ dst,
                         const _Float16* __restrict__ w1B,
                         const float* __restrict__ b1,
                         float* __restrict__ acc,
                         int n_tiles) {
    int lane = threadIdx.x & 31;
    int waveInBlock = threadIdx.x >> 5;
    int wavesPerBlock = blockDim.x >> 5;
    int wave = blockIdx.x * wavesPerBlock + waveInBlock;
    int nWaves = gridDim.x * wavesPerBlock;
    int m = lane & 15, hi = lane >> 4;
    int n = m;
    unsigned n4   = (unsigned)n * 4u;
    unsigned hi16 = (unsigned)hi * 16u;
    unsigned m8   = (unsigned)m * 8u;

    unsigned long long accB = (unsigned long long)(size_t)acc;
    const char* h16B = (const char*)h16;
    const char* srcB = (const char*)src;
    const char* dstB = (const char*)dst;

    // Weights resident in VGPRs across the whole grid-stride loop.
    v16h w1b[4];
    #pragma unroll
    for (int j = 0; j < 4; ++j)
        w1b[j] = *(const v16h*)(w1B + (j * 32 + lane) * 16);
    float b1v[4];
    #pragma unroll
    for (int j = 0; j < 4; ++j) b1v[j] = b1[j * 16 + n];

    // Prologue: indices for the first tile (low dword of int64 ids).
    int t0 = (wave < n_tiles) ? wave : 0;
    int srow = *(const int*)(srcB + ((unsigned)t0 * 128u + m8));
    int drow = *(const int*)(dstB + ((unsigned)t0 * 128u + m8));

    for (int t = wave; t < n_tiles; t += nWaves) {
        // A operand gather issues immediately (srow already resident).
        unsigned hoff = (unsigned)srow * 64u + hi16;
        v8h lo = *(const v8h*)(h16B + hoff);
        v8h hh = *(const v8h*)(h16B + hoff + 32u);
        v16h a = __builtin_shufflevector(lo, hh,
                 0,1,2,3,4,5,6,7,8,9,10,11,12,13,14,15);

        // dst row byte-offsets for C/D rows hi*8+v (drow already resident).
        unsigned doff[8];
        #pragma unroll
        for (int v = 0; v < 8; ++v) {
            int dro = __shfl(drow, hi * 8 + v, 32);
            doff[v] = ((unsigned)dro << 8) + n4;   // dst*64*4 + n*4
        }

        // Prefetch next tile's indices (uniform clamp -> branch-free, in-bounds).
        int t2 = t + nWaves;
        int tc = (t2 < n_tiles) ? t2 : t;
        unsigned eoff2 = (unsigned)tc * 128u + m8;
        int srow_n = *(const int*)(srcB + eoff2);
        int drow_n = *(const int*)(dstB + eoff2);

        // Issue all 4 independent WMMAs back-to-back, then consume.
        v8f d[4];
        #pragma unroll
        for (int j = 0; j < 4; ++j) {
            v8f c;
            #pragma unroll
            for (int v = 0; v < 8; ++v) c[v] = b1v[j];   // persistent bias regs
            d[j] = __builtin_amdgcn_wmma_f32_16x16x32_f16(
                       false, a, false, w1b[j], (short)0, c, false, false);
        }
        #pragma unroll
        for (int j = 0; j < 4; ++j)
            #pragma unroll
            for (int v = 0; v < 8; ++v)
                gfadd(doff[v], relu1(d[j][v]), accB, j * 64);

        srow = srow_n;
        drow = drow_n;
    }
}

// 4) Per-node layer-2: one wave = 16 nodes.
//    out = f16(acc)·W2 + deg⊗b2
__global__ void efn_out(const float* __restrict__ acc,
                        const float* __restrict__ deg,
                        const _Float16* __restrict__ w2B,
                        const float* __restrict__ b2,
                        float* __restrict__ out,
                        int n_tiles) {
    int lane = threadIdx.x & 31;
    int waveInBlock = threadIdx.x >> 5;
    int wavesPerBlock = blockDim.x >> 5;
    int wave = blockIdx.x * wavesPerBlock + waveInBlock;
    int nWaves = gridDim.x * wavesPerBlock;
    int m = lane & 15, hi = lane >> 4;
    int n = m;

    const char* accB = (const char*)acc;
    const char* degC = (const char*)deg;
    char* outB = (char*)out;

    v16h w2b[2][2];
    #pragma unroll
    for (int s = 0; s < 2; ++s)
        #pragma unroll
        for (int j = 0; j < 2; ++j)
            w2b[s][j] = *(const v16h*)(w2B + ((s * 2 + j) * 32 + lane) * 16);
    float b2v[2] = { b2[n], b2[16 + n] };

    for (int t = wave; t < n_tiles; t += nWaves) {
        unsigned nodeA = (unsigned)t * 16u + (unsigned)m;  // A-layout row
        v16h a[2];
        #pragma unroll
        for (int s = 0; s < 2; ++s) {
            const float* p =
                (const float*)(accB + nodeA * 256u + (unsigned)s * 128u
                               + (unsigned)hi * 32u);
            #pragma unroll
            for (int e = 0; e < 8; ++e) a[s][e]     = (_Float16)p[e];
            #pragma unroll
            for (int e = 0; e < 8; ++e) a[s][8 + e] = (_Float16)p[16 + e];
        }
        float degm = *(const float*)(degC + nodeA * 4u);
        float dg[8];
        #pragma unroll
        for (int v = 0; v < 8; ++v) dg[v] = __shfl(degm, hi * 8 + v, 32);

        v8f d[2];
        #pragma unroll
        for (int j = 0; j < 2; ++j) {
            v8f c;
            #pragma unroll
            for (int v = 0; v < 8; ++v) c[v] = dg[v] * b2v[j];
            c = __builtin_amdgcn_wmma_f32_16x16x32_f16(
                    false, a[0], false, w2b[0][j], (short)0, c, false, false);
            d[j] = __builtin_amdgcn_wmma_f32_16x16x32_f16(
                    false, a[1], false, w2b[1][j], (short)0, c, false, false);
        }
        #pragma unroll
        for (int j = 0; j < 2; ++j)
            #pragma unroll
            for (int v = 0; v < 8; ++v) {
                unsigned ooff = ((unsigned)t * 16u + (unsigned)(hi * 8 + v)) * 128u
                                + (unsigned)j * 64u + (unsigned)n * 4u;
                *(float*)(outB + ooff) = d[j][v];
            }
    }
}

extern "C" void kernel_launch(void* const* d_in, const int* in_sizes, int n_in,
                              void* d_out, int out_size, void* d_ws, size_t ws_size,
                              hipStream_t stream) {
    const float*      x       = (const float*)d_in[0];
    const float*      scalars = (const float*)d_in[1];
    const long long*  eidx    = (const long long*)d_in[2];   // int64 in reference
    const float*      W1      = (const float*)d_in[3];
    const float*      b1      = (const float*)d_in[4];
    const float*      W2      = (const float*)d_in[5];
    const float*      b2      = (const float*)d_in[6];
    float*            out     = (float*)d_out;

    // Workspace layout (bytes):
    //   h16 : 0          .. 6,400,000   (100000*32 f16)
    //   w1B : 6,400,000  .. +4096
    //   w2B : 6,404,096  .. +4096
    //   acc : 6,408,192  .. +25,600,000 (100000*64 f32)
    //   deg : 32,008,192 .. +400,000    (100000 f32, contiguous with acc)
    char* ws = (char*)d_ws;
    _Float16* h16 = (_Float16*)(ws);
    _Float16* w1B = (_Float16*)(ws + 6400000);
    _Float16* w2B = (_Float16*)(ws + 6404096);
    float*    acc = (float*)   (ws + 6408192);
    float*    deg = (float*)   (ws + 32008192);

    const long long* src = eidx;
    const long long* dst = eidx + N_EDGES;

    efn_zero<<<2048, 256, 0, stream>>>(acc, (long long)N_NODES * (H_DIM + 1));
    efn_prep_nodes<<<4096, 256, 0, stream>>>(x, scalars, h16);
    efn_prep_weights<<<1, 128, 0, stream>>>(W1, W2, w1B, w2B);
    efn_deg<<<2048, 256, 0, stream>>>(dst, deg, N_EDGES);
    efn_edge<<<2048, 256, 0, stream>>>(h16, src, dst, w1B, b1, acc,
                                       N_EDGES / 16);          // 200000 tiles
    efn_out<<<800, 256, 0, stream>>>(acc, deg, w2B, b2, out,
                                     N_NODES / 16);            // 6250 tiles
}
